// Segmenta_58875411694153
// MI455X (gfx1250) — matrix operand order repaired and checked by reading, compiled, not run on hardware
//
#include <hip/hip_runtime.h>
#include <hip/hip_bf16.h>

typedef __attribute__((ext_vector_type(16))) _Float16 v16h;
typedef __attribute__((ext_vector_type(8)))  _Float16 v8h;
typedef __attribute__((ext_vector_type(8)))  float    v8f;
typedef __attribute__((ext_vector_type(4)))  int      v4i;

union Frag16 { v16h v; v8h h[2]; };

#define AS1 __attribute__((address_space(1)))
#define AS3 __attribute__((address_space(3)))

// Prefer the CDNA5 async global->LDS path when the toolchain exposes it.
#if __has_builtin(__builtin_amdgcn_global_load_async_to_lds_b128)
#define HAVE_ASYNC_LDS 1
#else
#define HAVE_ASYNC_LDS 0
#endif

#if HAVE_ASYNC_LDS
#if __has_builtin(__builtin_amdgcn_s_wait_asynccnt)
#define WAIT_STAGE() __builtin_amdgcn_s_wait_asynccnt(0)
#else
#define WAIT_STAGE() asm volatile("s_wait_asynccnt 0x0" ::: "memory")
#endif
#else
#define WAIT_STAGE()
#endif

// ---------------------------------------------------------------------------
// Kernel A: f32 -> f16 weight conversion (grid-stride)
// ---------------------------------------------------------------------------
__global__ __launch_bounds__(256) void k_f32_to_f16(const float* __restrict__ src,
                                                    _Float16* __restrict__ dst,
                                                    long n) {
  long i = (long)blockIdx.x * blockDim.x + threadIdx.x;
  long stride = (long)gridDim.x * blockDim.x;
  for (; i < n; i += stride) dst[i] = (_Float16)src[i];
}

// ---------------------------------------------------------------------------
// Kernel B: per-ROI level select + ROI-align (7x7, SR=2) + CBAM.
// One block (256 threads = 8 waves) per ROI; thread = channel.
// Emits the flattened 12544-vector in f16 (GEMM A row, k = c*49 + p).
// ---------------------------------------------------------------------------
__global__ __launch_bounds__(256) void k_roi_cbam(
    const float* __restrict__ feat3, const float* __restrict__ feat4,
    const float* __restrict__ feat5,
    const float* __restrict__ boxes, const int* __restrict__ batch_idx,
    const float* __restrict__ ca_w1, const float* __restrict__ ca_w2,
    const float* __restrict__ sa_w,
    const int* __restrict__ img_h, const int* __restrict__ img_w,
    _Float16* __restrict__ crops_f16) {
  const int n = blockIdx.x;
  const int t = threadIdx.x;

  __shared__ float crop[256 * 49];            // stride 49 is coprime w/ 64 banks
  __shared__ int   xi0[14], xi1[14], yi0[14], yi1[14];
  __shared__ float lxv[14], vxa[14], lyv[14], vya[14];
  __shared__ float avgc[256], mxc[256];
  __shared__ float hid[32];
  __shared__ float smean[49], smax[49], saV[49];

  // --- box / level (computed redundantly per thread; hits L0) ---
  const float cx = boxes[n * 4 + 0], cy = boxes[n * 4 + 1];
  const float bw_ = boxes[n * 4 + 2], bh_ = boxes[n * 4 + 3];
  const float w_px = bw_ * (float)img_w[0];
  const float h_px = bh_ * (float)img_h[0];
  const float s = sqrtf(w_px * h_px + 1e-6f);
  const float lf = floorf(log2f(s / 56.f + 1e-6f));
  const int lvl = (int)fminf(fmaxf(lf, 0.f), 2.f);
  const float* feat; int fH, fW;
  if (lvl == 0)      { feat = feat3; fH = 64; fW = 64; }
  else if (lvl == 1) { feat = feat4; fH = 32; fW = 32; }
  else               { feat = feat5; fH = 16; fW = 16; }
  const int bi = batch_idx[n];

  const float x1 = (cx - 0.5f * bw_) * fW - 0.5f;
  const float x2 = (cx + 0.5f * bw_) * fW - 0.5f;
  const float y1 = (cy - 0.5f * bh_) * fH - 0.5f;
  const float y2 = (cy + 0.5f * bh_) * fH - 0.5f;
  const float binw = (x2 - x1) / 7.f;
  const float binh = (y2 - y1) / 7.f;

  // --- bilinear tables (threads 0..13: x, 16..29: y) ---
  if (t < 14) {
    float xs = x1 + binw * ((t + 0.5f) * 0.5f);
    vxa[t] = (xs >= -1.f && xs <= (float)fW) ? 1.f : 0.f;
    float xc = fminf(fmaxf(xs, 0.f), (float)(fW - 1));
    int x0 = (int)floorf(xc);
    xi0[t] = x0; xi1[t] = min(x0 + 1, fW - 1); lxv[t] = xc - (float)x0;
  } else if (t >= 16 && t < 30) {
    int i = t - 16;
    float ys = y1 + binh * ((i + 0.5f) * 0.5f);
    vya[i] = (ys >= -1.f && ys <= (float)fH) ? 1.f : 0.f;
    float yc = fminf(fmaxf(ys, 0.f), (float)(fH - 1));
    int y0 = (int)floorf(yc);
    yi0[i] = y0; yi1[i] = min(y0 + 1, fH - 1); lyv[i] = yc - (float)y0;
  }
  __syncthreads();

  // --- ROI-align: thread t = channel c ---
  const int c = t;
  const float* fp = feat + (long)(bi * 256 + c) * fH * fW;
  float asum = 0.f, amax = -INFINITY;
  for (int py = 0; py < 7; ++py) {
    for (int px = 0; px < 7; ++px) {
      float acc = 0.f;
#pragma unroll
      for (int sy = 0; sy < 2; ++sy) {
        const int iy = py * 2 + sy;
        const int ya = yi0[iy], yb = yi1[iy];
        const float ly = lyv[iy], vy = vya[iy];
#pragma unroll
        for (int sx = 0; sx < 2; ++sx) {
          const int ix = px * 2 + sx;
          const int xa = xi0[ix], xb = xi1[ix];
          const float lx = lxv[ix];
          const float v00 = fp[ya * fW + xa];
          const float v01 = fp[ya * fW + xb];
          const float v10 = fp[yb * fW + xa];
          const float v11 = fp[yb * fW + xb];
          float v = v00 * (1.f - ly) * (1.f - lx) + v01 * (1.f - ly) * lx +
                    v10 * ly * (1.f - lx) + v11 * ly * lx;
          acc += v * vy * vxa[ix];
        }
      }
      const float ov = acc * 0.25f;
      crop[c * 49 + py * 7 + px] = ov;
      asum += ov;
      amax = fmaxf(amax, ov);
    }
  }
  avgc[c] = asum * (1.f / 49.f);
  mxc[c] = amax;
  __syncthreads();

  // --- channel attention MLP: hidden (16) for avg path and max path ---
  if (t < 32) {
    const int hh = t & 15;
    const float* src = (t < 16) ? avgc : mxc;
    float acc = 0.f;
    for (int k = 0; k < 256; ++k) acc += ca_w1[hh * 256 + k] * src[k];
    hid[t] = fmaxf(acc, 0.f);
  }
  __syncthreads();

  float catt;
  {
    float acc = 0.f;
#pragma unroll
    for (int hh = 0; hh < 16; ++hh)
      acc += ca_w2[c * 16 + hh] * (hid[hh] + hid[16 + hh]);
    catt = 1.f / (1.f + expf(-acc));
  }
  for (int p = 0; p < 49; ++p) crop[c * 49 + p] *= catt;
  __syncthreads();

  // --- spatial stats across channels (threads 0..48 = pixel) ---
  if (t < 49) {
    float sm = 0.f, sx = -INFINITY;
    for (int k = 0; k < 256; ++k) {
      const float v = crop[k * 49 + t];
      sm += v; sx = fmaxf(sx, v);
    }
    smean[t] = sm * (1.f / 256.f);
    smax[t] = sx;
  }
  __syncthreads();

  // --- 7x7 spatial conv (pad 3) + sigmoid ---
  if (t < 49) {
    const int py = t / 7, px = t % 7;
    float acc = 0.f;
    for (int ky = 0; ky < 7; ++ky) {
      const int yy = py + ky - 3;
      if (yy < 0 || yy >= 7) continue;
      for (int kx = 0; kx < 7; ++kx) {
        const int xx = px + kx - 3;
        if (xx < 0 || xx >= 7) continue;
        acc += smean[yy * 7 + xx] * sa_w[ky * 7 + kx] +
               smax[yy * 7 + xx]  * sa_w[49 + ky * 7 + kx];
      }
    }
    saV[t] = 1.f / (1.f + expf(-acc));
  }
  __syncthreads();

  // --- emit f16 GEMM row ---
  _Float16* dst = crops_f16 + (long)n * 12544 + c * 49;
  for (int p = 0; p < 49; ++p)
    dst[p] = (_Float16)(crop[c * 49 + p] * saV[p]);
}

// ---------------------------------------------------------------------------
// Kernel C: fc1 GEMM via v_wmma_f32_16x16x32_f16, fused bias + ReLU.
// Block tile 64(M) x 128(N); 8 waves as 2x4; wave tile 32x32 -> 4 WMMA/step.
// Double-buffered LDS (one barrier per K-step); tiles staged with
// GLOBAL_LOAD_ASYNC_TO_LDS_B128 when available, else load+ds_store.
// LDS rows stride 40 halves (80 B) -> every fragment access is 16B-aligned.
// ---------------------------------------------------------------------------
#define GEMM_TM 64
#define GEMM_TN 128
#define GEMM_TK 32
#define LDS_STRIDE 40

__global__ __launch_bounds__(256) void k_gemm_fc1(
    const _Float16* __restrict__ A,      // [M][K] crops f16
    const _Float16* __restrict__ B,      // [Nw][K] weights f16 (row-major n,k)
    const float* __restrict__ bias,      // [Nw]
    float* __restrict__ C,               // [M][Nw] f32
    int M, int Nw, int K) {
  __shared__ alignas(16) _Float16 lA[2][GEMM_TM * LDS_STRIDE];
  __shared__ alignas(16) _Float16 lB[2][GEMM_TN * LDS_STRIDE];

  const int tid  = threadIdx.x;
  const int lane = tid & 31;
  const int wave = tid >> 5;
  const int mblk = blockIdx.x * GEMM_TM;
  const int nblk = blockIdx.y * GEMM_TN;
  const int m_off = (wave & 1) * 32;     // 2 waves along M, each owns 32 rows
  const int n_off = (wave >> 1) * 32;    // 4 waves along N, each owns 32 cols

  v8f acc[2][2] = {};

  // global->LDS staging indices
  const int ar  = tid >> 2;              // 0..63   (A row), one b128
  const int akc = (tid & 3) * 8;         // 0/8/16/24
  const int am  = mblk + ar;
  const int amc = am < M ? am : (M - 1); // clamp for async path (row discarded)
  const int bn  = tid >> 1;              // 0..127  (B row = out col), two b128
  const int bkc = (tid & 1) * 16;        // 0/16
  const long arow  = (long)am  * K;
  const long arowc = (long)amc * K;
  const long brow  = (long)(nblk + bn) * K;

  auto stage = [&](int kt, int b) {
    const int k0 = kt * GEMM_TK;
#if HAVE_ASYNC_LDS
    __builtin_amdgcn_global_load_async_to_lds_b128(
        (AS1 v4i*)(A + arowc + k0 + akc),
        (AS3 v4i*)&lA[b][ar * LDS_STRIDE + akc], 0, 0);
    __builtin_amdgcn_global_load_async_to_lds_b128(
        (AS1 v4i*)(B + brow + k0 + bkc),
        (AS3 v4i*)&lB[b][bn * LDS_STRIDE + bkc], 0, 0);
    __builtin_amdgcn_global_load_async_to_lds_b128(
        (AS1 v4i*)(B + brow + k0 + bkc + 8),
        (AS3 v4i*)&lB[b][bn * LDS_STRIDE + bkc + 8], 0, 0);
#else
    v8h av = {};
    if (am < M) av = *(const v8h*)(A + arow + k0 + akc);
    const v8h bv0 = *(const v8h*)(B + brow + k0 + bkc);
    const v8h bv1 = *(const v8h*)(B + brow + k0 + bkc + 8);
    *(v8h*)&lA[b][ar * LDS_STRIDE + akc] = av;
    *(v8h*)&lB[b][bn * LDS_STRIDE + bkc] = bv0;
    *(v8h*)&lB[b][bn * LDS_STRIDE + bkc + 8] = bv1;
    if (kt + 1 < K / GEMM_TK) {
      __builtin_prefetch(A + arow + k0 + GEMM_TK + akc, 0, 3);
      __builtin_prefetch(B + brow + k0 + GEMM_TK + bkc, 0, 3);
    }
#endif
  };

  // fragment indices (ISA §7.12.2 layouts)
  const int la = lane & 15;              // A: lane -> M row; B/D: lane -> N col
  const int lh = lane >> 4;

  const int nk = K / GEMM_TK;            // 12544 / 32 = 392
  stage(0, 0);

  for (int kt = 0; kt < nk; ++kt) {
    const int cur = kt & 1;
    WAIT_STAGE();                        // own async stores to LDS complete
    __syncthreads();                     // everyone's stores visible; prev reads done
    if (kt + 1 < nk) stage(kt + 1, cur ^ 1);

    Frag16 afr[2];
#pragma unroll
    for (int i = 0; i < 2; ++i) {
      const _Float16* base = &lA[cur][(m_off + 16 * i + la) * LDS_STRIDE + lh * 8];
      afr[i].h[0] = *(const v8h*)(base);        // K = kb..kb+7
      afr[i].h[1] = *(const v8h*)(base + 16);   // K = kb+16..kb+23
    }
#pragma unroll
    for (int j = 0; j < 2; ++j) {
      Frag16 bfr;
      const _Float16* base = &lB[cur][(n_off + 16 * j + la) * LDS_STRIDE + lh * 8];
      bfr.h[0] = *(const v8h*)(base);           // K = kb..kb+7
      bfr.h[1] = *(const v8h*)(base + 16);
#pragma unroll
      for (int i = 0; i < 2; ++i)
        acc[i][j] = __builtin_amdgcn_wmma_f32_16x16x32_f16(
            /*neg_a=*/false, afr[i].v, /*neg_b=*/false, bfr.v,
            /*c_mod=*/(short)0, acc[i][j], /*reuse_a=*/false, /*reuse_b=*/false);
    }
  }

  // store with fused bias + ReLU; D layout: VGPR r -> M = r + 8*(lane/16)
#pragma unroll
  for (int j = 0; j < 2; ++j) {
    const int ncol = nblk + n_off + 16 * j + la;
    const float bv = bias[ncol];
#pragma unroll
    for (int i = 0; i < 2; ++i) {
#pragma unroll
      for (int r = 0; r < 8; ++r) {
        const int m = mblk + m_off + 16 * i + lh * 8 + r;
        if (m < M) {
          const float v = acc[i][j][r] + bv;
          C[(long)m * Nw + ncol] = v > 0.f ? v : 0.f;
        }
      }
    }
  }
}

// ---------------------------------------------------------------------------
// Kernel D: fc2 (1024 -> 5) with LDS tree reduction; one block per ROI.
// ---------------------------------------------------------------------------
__global__ __launch_bounds__(256) void k_fc2(const float* __restrict__ h1,
                                             const float* __restrict__ w2,
                                             const float* __restrict__ b2,
                                             float* __restrict__ out) {
  const int n = blockIdx.x;
  const int t = threadIdx.x;
  __shared__ float red[256];
  float acc[5] = {0.f, 0.f, 0.f, 0.f, 0.f};
  for (int k = t; k < 1024; k += 256) {
    const float hv = h1[(long)n * 1024 + k];
#pragma unroll
    for (int j = 0; j < 5; ++j) acc[j] += hv * w2[j * 1024 + k];
  }
#pragma unroll
  for (int j = 0; j < 5; ++j) {
    red[t] = acc[j];
    __syncthreads();
    for (int off = 128; off > 0; off >>= 1) {
      if (t < off) red[t] += red[t + off];
      __syncthreads();
    }
    if (t == 0) out[(long)n * 5 + j] = red[0] + b2[j];
    __syncthreads();
  }
}

// ---------------------------------------------------------------------------
extern "C" void kernel_launch(void* const* d_in, const int* in_sizes, int n_in,
                              void* d_out, int out_size, void* d_ws, size_t ws_size,
                              hipStream_t stream) {
  (void)n_in; (void)out_size; (void)ws_size;
  const float* feat3  = (const float*)d_in[0];
  const float* feat4  = (const float*)d_in[1];
  const float* feat5  = (const float*)d_in[2];
  const float* boxes  = (const float*)d_in[3];
  const int*   bidx   = (const int*)d_in[4];
  const float* ca_w1  = (const float*)d_in[5];
  const float* ca_w2  = (const float*)d_in[6];
  const float* sa_w   = (const float*)d_in[7];
  const float* fc1_w  = (const float*)d_in[8];
  const float* fc1_b  = (const float*)d_in[9];
  const float* fc2_w  = (const float*)d_in[10];
  const float* fc2_b  = (const float*)d_in[11];
  const int*   img_h  = (const int*)d_in[12];
  const int*   img_w  = (const int*)d_in[13];

  const int N  = in_sizes[3] / 4;   // 1000 ROIs
  const int K  = 12544;             // 256 * 7 * 7
  const int H1 = 1024;

  // workspace layout (256B aligned regions)
  char* ws = (char*)d_ws;
  const size_t sz_crops = (size_t)N * K * sizeof(_Float16);
  const size_t off_w    = (sz_crops + 255) & ~(size_t)255;
  const size_t sz_w     = (size_t)H1 * K * sizeof(_Float16);
  const size_t off_h1   = (off_w + sz_w + 255) & ~(size_t)255;

  _Float16* crops = (_Float16*)ws;
  _Float16* wgt   = (_Float16*)(ws + off_w);
  float*    h1    = (float*)(ws + off_h1);

  // A: weight conversion
  k_f32_to_f16<<<1024, 256, 0, stream>>>(fc1_w, wgt, (long)H1 * K);
  // B: ROI-align + CBAM -> f16 activation rows
  k_roi_cbam<<<N, 256, 0, stream>>>(feat3, feat4, feat5, boxes, bidx,
                                    ca_w1, ca_w2, sa_w, img_h, img_w, crops);
  // C: fc1 GEMM (WMMA) + bias + ReLU
  dim3 g((N + GEMM_TM - 1) / GEMM_TM, H1 / GEMM_TN);
  k_gemm_fc1<<<g, 256, 0, stream>>>(crops, wgt, fc1_b, h1, N, H1, K);
  // D: fc2
  k_fc2<<<N, 256, 0, stream>>>(h1, fc2_w, fc2_b, (float*)d_out);
}
